// Custom_Model_38397007626975
// MI455X (gfx1250) — matrix operand order repaired and checked by reading, not compiled, not run on hardware
//
#include <hip/hip_runtime.h>
#include <hip/hip_bf16.h>

#define N_NODES  50000
#define M_PAD    50048   // N_NODES rounded up to 64: all GEMM tiles are full
#define N_EDGES  800000
#define N_GRAPHS 512
#define HID      128
#define NG       50
#define NGP      64     // padded RBF K
#define NLAYER   6
#define EDGE_ITERS 4    // 4 x 64 = 256 edges per block, weights staged once
#define CUTOFF_F 10.0f
#define PI_F     3.14159265358979f
#define LOG2_F   0.6931471805599453f   // ln(2)
#define LOG2E_F  1.4426950408889634f   // log2(e)
#define RBF_DELTA (CUTOFF_F / (NG - 1))
#define RBF_COEFF (-0.5f / (RBF_DELTA * RBF_DELTA))

typedef __attribute__((ext_vector_type(16))) _Float16 v16h;
typedef __attribute__((ext_vector_type(8)))  float    v8f;
typedef __attribute__((ext_vector_type(4)))  unsigned u32x4;
typedef __attribute__((ext_vector_type(4)))  int      i32x4;
typedef __attribute__((ext_vector_type(8)))  int      i32x8;

#if __has_builtin(__builtin_amdgcn_tensor_load_to_lds)
#define HAVE_TDM 1
#else
#define HAVE_TDM 0
#endif

__device__ __forceinline__ float fast_exp2(float x) {
#if __has_builtin(__builtin_amdgcn_exp2f)
  return __builtin_amdgcn_exp2f(x);      // bare v_exp_f32
#else
  return __expf(x * LOG2_F);
#endif
}
__device__ __forceinline__ float fast_log2(float x) {
#if __has_builtin(__builtin_amdgcn_logf)
  return __builtin_amdgcn_logf(x);       // bare v_log_f32
#else
  return __log2f(x);
#endif
}

__device__ __forceinline__ float ssp_f(float x) {
  // softplus(x) - ln2, in log2 space with raw HW transcendentals:
  //   t = 2^(-|x|*log2e) = e^{-|x|}  in (0,1]  ->  1+t in (1,2], log2 well-conditioned
  float t = fast_exp2(-fabsf(x) * LOG2E_F);
  return fmaxf(x, 0.f) + fmaf(LOG2_F, fast_log2(1.0f + t), -LOG2_F);
}

// ---------------------------------------------------------------- WMMA fragments
// A fragment: 16x32 (MxK) f16 from row-major LDS. Lane l<16: row l, K 0..15;
// lane l>=16: row l-16, K 16..31. Lowers to 2x ds_load_b128.
__device__ __forceinline__ v16h load_a_frag(const _Float16* A, int lda, int row0, int k0, int lane) {
  const int hh = lane >> 4;
  const _Float16* p = A + (row0 + (lane & 15)) * lda + k0 + hh * 16;
  v16h a;
#pragma unroll
  for (int e = 0; e < 16; ++e) a[e] = p[e];
  return a;
}

// B fragment from pre-swizzled (fragment-major) storage: lane's 16 values are 32
// contiguous bytes -> 2x ds_load_b128, no packing VALU.
__device__ __forceinline__ v16h load_b_frag_sw(const _Float16* sB, int frag, int lane) {
  const _Float16* p = sB + ((frag << 5) + lane) * 16;
  v16h b;
#pragma unroll
  for (int e = 0; e < 16; ++e) b[e] = p[e];
  return b;
}

// ---------------------------------------------------------------- TDM 1-D load
// Loads nelem f16 elements from global (contiguous) into LDS byte offset lds_off.
// D# packed per ISA 8.3/8.4: group0 = {count=1, lds_addr, global_addr, type=2},
// group1 = {data_size=2B, tensor_dim0=tile_dim0=nelem}. Dynamic-LDS region assumed
// to start at LDS offset 0 (no static LDS in these kernels).
__device__ __forceinline__ void tdm_load_1d(unsigned lds_off, const void* gsrc, unsigned nelem) {
#if HAVE_TDM
  unsigned long long ga = (unsigned long long)(uintptr_t)gsrc;
  u32x4 g0;
  g0[0] = 1u;                                                  // count=1 (valid user D#)
  g0[1] = lds_off;                                             // lds_addr (bytes)
  g0[2] = (unsigned)ga;                                        // global_addr[31:0]
  g0[3] = (unsigned)((ga >> 32) & 0x1FFFFFFu) | (2u << 30);    // global_addr[56:32] | type=2
  i32x8 g1;
  g1[0] = 0x00010000;                                          // data_size=1 (2 bytes)
  g1[1] = (int)((nelem & 0xFFFFu) << 16);                      // tensor_dim0[15:0]
  g1[2] = (int)((nelem >> 16) & 0xFFFFu);                      // tensor_dim0[31:16]
  g1[3] = (int)((nelem & 0xFFFFu) << 16);                      // tile_dim0 (<=65535)
  g1[4] = 0;                                                   // tile_dim1/2 unused
  g1[5] = (int)nelem;                                          // tensor_dim0_stride[31:0]
  g1[6] = 0;
  g1[7] = 0;
  i32x4 gz = {0, 0, 0, 0};
#if __has_include(<hip/amd_detail/amd_gfx1250_TDM.h>)
  i32x8 gz8 = {0, 0, 0, 0, 0, 0, 0, 0};
  __builtin_amdgcn_tensor_load_to_lds(g0, g1, gz, gz, gz8, 0); // clang-23 6-arg form
#else
  __builtin_amdgcn_tensor_load_to_lds(g0, g1, gz, gz, 0);      // ROCm 7.2 5-arg form
#endif
#endif
}

__device__ __forceinline__ void stage_copy16(_Float16* dst, const _Float16* src, int nhalf, int tid) {
  // fallback flat staging copy, 16B per op
  const uint4* s = (const uint4*)src;
  uint4* d = (uint4*)dst;
  for (int i = tid; i < (nhalf >> 3); i += 256) d[i] = s[i];
}

// ---------------------------------------------------------------- utility kernels
__global__ void zero_f32_kernel(float* p, int n) {
  int i = blockIdx.x * blockDim.x + threadIdx.x;
  if (i < n) p[i] = 0.f;
}

__global__ void init_v_kernel(const int* z, const float* emb_w, float* v) {
  int i = blockIdx.x * blockDim.x + threadIdx.x;
  if (i < M_PAD * HID) {
    int n = i >> 7, c = i & 127;
    v[i] = (n < N_NODES) ? emb_w[z[n] * HID + c] : 0.f;
  }
}

__global__ void edge_geom_kernel(const float* pos, const int* row, const int* col,
                                 float* dist, float* cc) {
  int i = blockIdx.x * blockDim.x + threadIdx.x;
  if (i < N_EDGES) {
    int r = row[i], c = col[i];
    float dx = pos[r * 3 + 0] - pos[c * 3 + 0];
    float dy = pos[r * 3 + 1] - pos[c * 3 + 1];
    float dz = pos[r * 3 + 2] - pos[c * 3 + 2];
    float d = sqrtf(dx * dx + dy * dy + dz * dz + 1e-12f);
    dist[i] = d;
    cc[i] = 0.5f * (__cosf(d * PI_F / CUTOFF_F) + 1.f);
  }
}

// Re-pack a weight matrix Wt[N][Ksrc] (row-major, f32) into fragment-major f16:
// dst[((kc*ntn + nt)*32 + lane)*16 + e] = Wt[nt*16 + (lane&15)][kc*32 + (lane>>4)*16 + e]
// zero-padded for k >= Ksrc. K multiple of 32, N multiple of 16.
__global__ void swizzle_weight_kernel(const float* Wt, int Ksrc, int K, int N, _Float16* dst) {
  int i = blockIdx.x * blockDim.x + threadIdx.x;
  if (i < K * N) {
    int frag = i >> 9;
    int lane = (i >> 4) & 31;
    int e = i & 15;
    int ntn = N >> 4;
    int kc = frag / ntn, nt = frag % ntn;
    int k = kc * 32 + (lane >> 4) * 16 + e;
    int n = nt * 16 + (lane & 15);
    dst[i] = (k < Ksrc) ? (_Float16)Wt[n * Ksrc + k] : (_Float16)0.f;
  }
}

// ---------------------------------------------------------------- node GEMM (WMMA)
// out[M_PAD,N] = epilogue(A[M_PAD,128] @ W^T). All tiles full (M padded to 64).
// MODE 0: outh = (f16)acc          MODE 1: outh = (f16)ssp(acc+bias)
// MODE 2: outf = resid + acc + bias (f32)
template <int N, int MODE, int AHALF>
__global__ void wmma_gemm_node(const void* __restrict__ Av,
                               const _Float16* __restrict__ Bsw, const float* __restrict__ bias,
                               const float* __restrict__ resid,
                               _Float16* __restrict__ outh, float* __restrict__ outf) {
  extern __shared__ char smem[];
  _Float16* sA = (_Float16*)smem;            // 64*128 halfs @ LDS off 0
  _Float16* sB = sA + 64 * HID;              // 128*N halfs  @ LDS off 16384
  float* sbias = (float*)(sB + HID * N);

  const int tid = threadIdx.x, lane = tid & 31, wave = tid >> 5;
  const int m0 = blockIdx.x * 64;

#if HAVE_TDM
  if (wave == 0) {
    tdm_load_1d(64 * HID * 2, Bsw, (unsigned)(HID * N));
    __builtin_amdgcn_s_wait_tensorcnt(0);
  }
#else
  stage_copy16(sB, Bsw, HID * N, tid);
#endif
  if (MODE != 0 && tid < N) sbias[tid] = bias[tid];

  // A staging: 4 contiguous elements per thread -> b128 loads / b64 LDS stores
  for (int i = tid; i < 64 * HID / 4; i += 256) {
    int r = i >> 5, c4 = (i & 31) << 2;
    int gr = m0 + r;
    _Float16* d = sA + r * HID + c4;
    if (AHALF) {
      const _Float16* p = (const _Float16*)Av + (size_t)gr * HID + c4;
      d[0] = p[0]; d[1] = p[1]; d[2] = p[2]; d[3] = p[3];
    } else {
      const float4 f = *(const float4*)((const float*)Av + (size_t)gr * HID + c4);
      d[0] = (_Float16)f.x; d[1] = (_Float16)f.y; d[2] = (_Float16)f.z; d[3] = (_Float16)f.w;
    }
  }
  __syncthreads();

  constexpr int NTN = N >> 4;
  constexpr int NTASK_PER_WAVE = (4 * NTN) / 8;
#pragma unroll
  for (int t0 = 0; t0 < NTASK_PER_WAVE; ++t0) {
    const int t = wave + t0 * 8;
    const int et = t / NTN, nt = t % NTN;
    v8f acc = {};
#pragma unroll
    for (int kc = 0; kc < 4; ++kc) {
      v16h a = load_a_frag(sA, HID, et * 16, kc * 32, lane);
      v16h b = load_b_frag_sw(sB, kc * NTN + nt, lane);
      acc = __builtin_amdgcn_wmma_f32_16x16x32_f16(false, a, false, b, (short)0, acc, false, false);
    }
    const int colb = nt * 16 + (lane & 15);
    const size_t base = (size_t)(m0 + et * 16 + (lane >> 4) * 8) * N + colb;
    const float bb = (MODE != 0) ? sbias[colb] : 0.f;
    // single base pointer per task; r*N folds into the store immediate offset
    if (MODE == 2) {
      float* op = outf + base;
      const float* rp = resid + base;
#pragma unroll
      for (int r = 0; r < 8; ++r) op[r * N] = rp[r * N] + acc[r] + bb;
    } else {
      _Float16* op = outh + base;
#pragma unroll
      for (int r = 0; r < 8; ++r)
        op[r * N] = (MODE == 0) ? (_Float16)acc[r] : (_Float16)ssp_f(acc[r] + bb);
    }
  }
}

// ---------------------------------------------------------------- fused edge kernel
// Per block (256 thr, 8 waves): 256 edges (4 x 64), weights staged once via TDM.
//   t  = ssp(de @ m0w^T + b0)   [64,128]  (WMMA, K=64 padded)
//   W  = (t @ m2w^T + b2) * C   [64,128]  (WMMA, K=128)
//   agg[col] += xW[row] * W     (gather + f32 atomics)
__global__ void edge_mp_kernel(const float* __restrict__ dist, const float* __restrict__ cc,
                               const int* __restrict__ row, const int* __restrict__ col,
                               const _Float16* __restrict__ B0sw,  // swizzled [NGP x 128]
                               const float* __restrict__ m0b,      // [128]
                               const _Float16* __restrict__ B2sw,  // swizzled [128 x 128]
                               const float* __restrict__ m2b,      // [128]
                               const _Float16* __restrict__ xW,    // [M_PAD,128] f16
                               float* __restrict__ agg) {          // [M_PAD,128] f32
  extern __shared__ char smem[];
  _Float16* sB0 = (_Float16*)smem;          // 64*128   @ LDS off 0
  _Float16* sB2 = sB0 + NGP * HID;          // 128*128  @ LDS off 16384
  _Float16* sDE = sB2 + HID * HID;          // [64][64] rbf, K-padded
  _Float16* sT  = sDE + 64 * NGP;           // [64][128] hidden (A of GEMM2)
  float* sW     = (float*)(sT + 64 * HID);  // [64][128] final filter
  float* sDist  = sW + 64 * HID;            // [64]
  float* sC     = sDist + 64;               // [64]
  float* sBias0 = sC + 64;                  // [128]
  float* sBias2 = sBias0 + HID;             // [128]

  const int tid = threadIdx.x, lane = tid & 31, wave = tid >> 5;

  // stage weights once per block (TDM bulk DMA into LDS)
#if HAVE_TDM
  if (wave == 0) {
    tdm_load_1d(0u, B0sw, NGP * HID);
    tdm_load_1d((unsigned)(NGP * HID * 2), B2sw, HID * HID);
    __builtin_amdgcn_s_wait_tensorcnt(0);
  }
#else
  stage_copy16(sB0, B0sw, NGP * HID, tid);
  stage_copy16(sB2, B2sw, HID * HID, tid);
#endif
  if (tid < HID) { sBias0[tid] = m0b[tid]; sBias2[tid] = m2b[tid]; }

#pragma unroll 1
  for (int it = 0; it < EDGE_ITERS; ++it) {
    const int eb = (blockIdx.x * EDGE_ITERS + it) * 64;   // N_EDGES % (64*EDGE_ITERS) == 0

    if (tid < 64) {
      sDist[tid] = dist[eb + tid];
      sC[tid] = cc[eb + tid];
      // warm the gather rows while we do the edge MLP (global_prefetch_b8)
      __builtin_prefetch(xW + (size_t)row[eb + tid] * HID, 0, 1);
    }
    __syncthreads();
    for (int idx = tid; idx < 64 * NGP; idx += 256) {
      int e = idx >> 6, g = idx & 63;
      float val = 0.f;
      if (g < NG) {
        float u = sDist[e] - (float)g * RBF_DELTA;
        val = fast_exp2(RBF_COEFF * LOG2E_F * u * u);
      }
      sDE[idx] = (_Float16)val;
    }
    __syncthreads();

    // stage 1: t = ssp(de @ m0w^T + b0)   -- 4 edge-tiles x 8 col-tiles, K=64
#pragma unroll
    for (int t0 = 0; t0 < 4; ++t0) {
      const int t = wave + t0 * 8;
      const int et = t >> 3, nt = t & 7;
      v8f acc = {};
#pragma unroll
      for (int kc = 0; kc < 2; ++kc) {
        v16h a = load_a_frag(sDE, NGP, et * 16, kc * 32, lane);
        v16h b = load_b_frag_sw(sB0, kc * 8 + nt, lane);
        acc = __builtin_amdgcn_wmma_f32_16x16x32_f16(false, a, false, b, (short)0, acc, false, false);
      }
      const int colb = nt * 16 + (lane & 15);
      const float bb = sBias0[colb];
      _Float16* tp = sT + (et * 16 + (lane >> 4) * 8) * HID + colb;  // r*HID folds into ds offset
#pragma unroll
      for (int r = 0; r < 8; ++r) tp[r * HID] = (_Float16)ssp_f(acc[r] + bb);
    }
    __syncthreads();

    // stage 2: W = (t @ m2w^T + b2) * C   -- K=128
#pragma unroll
    for (int t0 = 0; t0 < 4; ++t0) {
      const int t = wave + t0 * 8;
      const int et = t >> 3, nt = t & 7;
      v8f acc = {};
#pragma unroll
      for (int kc = 0; kc < 4; ++kc) {
        v16h a = load_a_frag(sT, HID, et * 16, kc * 32, lane);
        v16h b = load_b_frag_sw(sB2, kc * 8 + nt, lane);
        acc = __builtin_amdgcn_wmma_f32_16x16x32_f16(false, a, false, b, (short)0, acc, false, false);
      }
      const int colb = nt * 16 + (lane & 15);
      const float bb = sBias2[colb];
      const int e0 = et * 16 + (lane >> 4) * 8;
      const float* cp = sC + e0;                   // 8 consecutive floats
      float* wp = sW + e0 * HID + colb;            // r*HID folds into ds offset
#pragma unroll
      for (int r = 0; r < 8; ++r) wp[r * HID] = (acc[r] + bb) * cp[r];
    }
    __syncthreads();

    // stage 3: gather xW[row] (4 cols/thread), multiply, scatter-add into agg[col]
    for (int idx = tid; idx < 64 * HID / 4; idx += 256) {
      int e = idx >> 5, c4 = (idx & 31) << 2;
      int ge = eb + e;
      const _Float16* xp = xW + (size_t)row[ge] * HID + c4;
      float* ap = agg + (size_t)col[ge] * HID + c4;
      const float* wp = sW + e * HID + c4;
#pragma unroll
      for (int u = 0; u < 4; ++u) atomicAdd(ap + u, wp[u] * (float)xp[u]);
    }
    __syncthreads();  // protect sDist/sDE/sT/sW before next iteration
  }
}

// ---------------------------------------------------------------- readout
__global__ void readout_kernel(const _Float16* __restrict__ hh,  // [M_PAD,64]
                               const float* __restrict__ u2w,    // [64]
                               const float* __restrict__ u2b,    // [1]
                               const int* __restrict__ batch,
                               float* __restrict__ out) {        // [512]
  int n = blockIdx.x * blockDim.x + threadIdx.x;
  if (n < N_NODES) {
    const _Float16* h = hh + (size_t)n * 64;
    float s = u2b[0];
#pragma unroll
    for (int c = 0; c < 64; ++c) s += (float)h[c] * u2w[c];
    atomicAdd(&out[batch[n]], s);
  }
}

// ---------------------------------------------------------------- launch
extern "C" void kernel_launch(void* const* d_in, const int* in_sizes, int n_in,
                              void* d_out, int out_size, void* d_ws, size_t ws_size,
                              hipStream_t stream) {
  const int*   z      = (const int*)d_in[0];
  const float* pos    = (const float*)d_in[1];
  const int*   batch  = (const int*)d_in[2];
  const int*   eidx   = (const int*)d_in[3];
  const float* emb_w  = (const float*)d_in[4];
  const float* lin_w  = (const float*)d_in[5];
  const float* mlp0_w = (const float*)d_in[6];
  const float* mlp0_b = (const float*)d_in[7];
  const float* mlp2_w = (const float*)d_in[8];
  const float* mlp2_b = (const float*)d_in[9];
  const float* v1_w   = (const float*)d_in[10];
  const float* v1_b   = (const float*)d_in[11];
  const float* v2_w   = (const float*)d_in[12];
  const float* v2_b   = (const float*)d_in[13];
  const float* u1_w   = (const float*)d_in[14];
  const float* u1_b   = (const float*)d_in[15];
  const float* u2_w   = (const float*)d_in[16];
  const float* u2_b   = (const float*)d_in[17];
  const int* row = eidx;
  const int* col = eidx + N_EDGES;
  float* out = (float*)d_out;

  // workspace carve-up (256B aligned); node buffers padded to M_PAD rows
  char* ws = (char*)d_ws;
  size_t off = 0;
  auto carve = [&](size_t bytes) { char* p = ws + off; off = (off + bytes + 255) & ~(size_t)255; return p; };
  float*    dist = (float*)carve((size_t)N_EDGES * 4);
  float*    ccut = (float*)carve((size_t)N_EDGES * 4);
  float*    v    = (float*)carve((size_t)M_PAD * HID * 4);
  float*    agg  = (float*)carve((size_t)M_PAD * HID * 4);
  _Float16* xWh  = (_Float16*)carve((size_t)M_PAD * HID * 2);
  _Float16* h1h  = (_Float16*)carve((size_t)M_PAD * HID * 2);
  _Float16* hhh  = (_Float16*)carve((size_t)M_PAD * 64 * 2);
  // swizzled f16 weights: per layer B0(64x128) + B2,lw,w1,w2(128x128), + u1(128x64)
  const size_t LSTRIDE = (size_t)NGP * HID + 4 * (size_t)HID * HID;  // 73728 halfs
  _Float16* wsw = (_Float16*)carve((6 * LSTRIDE + (size_t)HID * 64) * 2);
  (void)ws_size; (void)n_in; (void)in_sizes; (void)out_size;

  const size_t gemm_lds = 64 * HID * 2 + HID * HID * 2 + HID * 4;      // 49664 B
  const size_t gemm_lds64 = 64 * HID * 2 + HID * 64 * 2 + 64 * 4;
  const size_t edge_lds = (NGP * HID + HID * HID + 64 * NGP + 64 * HID) * 2
                        + (64 * HID + 64 + 64 + HID + HID) * 4;        // 108032 B
  const int gemm_grid = M_PAD / 64;                                    // 782, all full tiles
  const int edge_grid = N_EDGES / (64 * EDGE_ITERS);

  // ---- weight pre-swizzle (fragment-major f16) ----
  for (int l = 0; l < NLAYER; ++l) {
    _Float16* B0s = wsw + l * LSTRIDE;
    _Float16* B2s = B0s + (size_t)NGP * HID;
    _Float16* lws = B2s + (size_t)HID * HID;
    _Float16* w1s = lws + (size_t)HID * HID;
    _Float16* w2s = w1s + (size_t)HID * HID;
    swizzle_weight_kernel<<<(NGP * HID + 255) / 256, 256, 0, stream>>>(
        mlp0_w + (size_t)l * HID * NG, NG, NGP, HID, B0s);
    swizzle_weight_kernel<<<(HID * HID + 255) / 256, 256, 0, stream>>>(
        mlp2_w + (size_t)l * HID * HID, HID, HID, HID, B2s);
    swizzle_weight_kernel<<<(HID * HID + 255) / 256, 256, 0, stream>>>(
        lin_w + (size_t)l * HID * HID, HID, HID, HID, lws);
    swizzle_weight_kernel<<<(HID * HID + 255) / 256, 256, 0, stream>>>(
        v1_w + (size_t)l * HID * HID, HID, HID, HID, w1s);
    swizzle_weight_kernel<<<(HID * HID + 255) / 256, 256, 0, stream>>>(
        v2_w + (size_t)l * HID * HID, HID, HID, HID, w2s);
  }
  _Float16* u1s = wsw + 6 * LSTRIDE;
  swizzle_weight_kernel<<<(HID * 64 + 255) / 256, 256, 0, stream>>>(u1_w, HID, HID, 64, u1s);

  edge_geom_kernel<<<(N_EDGES + 255) / 256, 256, 0, stream>>>(pos, row, col, dist, ccut);
  init_v_kernel<<<(M_PAD * HID + 255) / 256, 256, 0, stream>>>(z, emb_w, v);

  for (int l = 0; l < NLAYER; ++l) {
    _Float16* B0s = wsw + l * LSTRIDE;
    _Float16* B2s = B0s + (size_t)NGP * HID;
    _Float16* lws = B2s + (size_t)HID * HID;
    _Float16* w1s = lws + (size_t)HID * HID;
    _Float16* w2s = w1s + (size_t)HID * HID;
    const float* m0b = mlp0_b + (size_t)l * HID;
    const float* m2b = mlp2_b + (size_t)l * HID;
    const float* b1  = v1_b + (size_t)l * HID;
    const float* b2  = v2_b + (size_t)l * HID;

    // xW = v @ lw^T  (f16 out, no bias)
    wmma_gemm_node<HID, 0, 0><<<gemm_grid, 256, gemm_lds, stream>>>(
        v, lws, nullptr, nullptr, xWh, nullptr);
    zero_f32_kernel<<<(M_PAD * HID + 255) / 256, 256, 0, stream>>>(agg, M_PAD * HID);
    edge_mp_kernel<<<edge_grid, 256, edge_lds, stream>>>(dist, ccut, row, col,
                                                         B0s, m0b, B2s, m2b, xWh, agg);
    // h1 = ssp(agg @ w1^T + b1)  (f16 out)
    wmma_gemm_node<HID, 1, 0><<<gemm_grid, 256, gemm_lds, stream>>>(
        agg, w1s, b1, nullptr, h1h, nullptr);
    // v = v + h1 @ w2^T + b2  (f32, in-place residual)
    wmma_gemm_node<HID, 2, 1><<<gemm_grid, 256, gemm_lds, stream>>>(
        h1h, w2s, b2, v, nullptr, v);
  }

  // readout: hh = ssp(v @ u1^T + b1)  [M_PAD,64]
  wmma_gemm_node<64, 1, 0><<<gemm_grid, 256, gemm_lds64, stream>>>(
      v, u1s, u1_b, nullptr, hhh, nullptr);
  zero_f32_kernel<<<(N_GRAPHS + 255) / 256, 256, 0, stream>>>(out, N_GRAPHS);
  readout_kernel<<<(N_NODES + 255) / 256, 256, 0, stream>>>(hhh, u2_w, u2_b, batch, out);
}
